// Qwen3Model_82386062672470
// MI455X (gfx1250) — matrix operand order repaired and checked by reading, not compiled, not run on hardware
//
#include <hip/hip_runtime.h>
#include <hip/hip_bf16.h>

// ---------------------------------------------------------------------------
// Qwen3-style decoder forward for MI455X (gfx1250, wave32, WMMA bf16).
//
// - All GEMMs (QKV, Wo, gate/up/down, lm_head) and both attention matmuls run
//   on v_wmma_f32_16x16x32_bf16 with fp32 accumulation (compute-bound:
//   ~1000 FLOP/weight-byte at M=2048, far above the 23.3 TB/s HBM knee).
// - Activation (A) matrices are produced in bf16 by their producer kernels,
//   so the GEMM stages A with GLOBAL_LOAD_ASYNC_TO_LDS_B128 (CDNA5 async
//   copy, tracked by ASYNCcnt) into a double-buffered LDS tile while the
//   WMMA pipe computes the previous tile; one s_wait_asynccnt + barrier per
//   K-step. Weights stay fp32 and go through load->cvt->transposed ds_store
//   (transpose is not expressible as 16B async chunks).
// ---------------------------------------------------------------------------

typedef __attribute__((ext_vector_type(16))) __bf16 v16bf;
typedef __attribute__((ext_vector_type(8)))  __bf16 v8bf;
typedef __attribute__((ext_vector_type(4)))  __bf16 v4bf;
typedef __attribute__((ext_vector_type(8)))  float  v8f;

static constexpr int Bc   = 2;
static constexpr int Sc   = 1024;
static constexpr int Hc   = 1024;
static constexpr int NHc  = 16;
static constexpr int NKVc = 8;
static constexpr int HDc  = 64;
static constexpr int Lc   = 8;
static constexpr int Ic   = 4096;
static constexpr int Vc   = 32000;

// Build a 16-element bf16 A/B fragment from two aligned 16-byte LDS chunks.
__device__ __forceinline__ v16bf ld_frag(const __bf16* p0, const __bf16* p1) {
  v8bf lo = *(const v8bf*)p0;
  v8bf hi = *(const v8bf*)p1;
  return __builtin_shufflevector(lo, hi, 0,1,2,3,4,5,6,7,8,9,10,11,12,13,14,15);
}

__device__ __forceinline__ v8f wmma_bf16(v16bf a, v16bf b, v8f c) {
  // (neg_a, A, neg_b, B, c_mod, C, reuse_a, reuse_b)
  return __builtin_amdgcn_wmma_f32_16x16x32_bf16(false, a, false, b,
                                                 (short)0, c, false, false);
}

// CDNA5 async copy: 16 bytes global -> LDS, tracked by ASYNCcnt.
// VDST = per-lane LDS byte offset (low 32 bits of the flat shared pointer,
// ISA 10.2), VADDR = per-lane 64-bit global address.
__device__ __forceinline__ void async_ld16(const __bf16* lds_ptr,
                                           const __bf16* gptr) {
  const unsigned       lds = (unsigned)(unsigned long long)lds_ptr;
  const unsigned long long g = (unsigned long long)gptr;
  asm volatile("global_load_async_to_lds_b128 %0, %1, off"
               :: "v"(lds), "v"(g) : "memory");
}

__device__ __forceinline__ void wait_async0() {
  asm volatile("s_wait_asynccnt 0x0" ::: "memory");
}

// ---------------------------------------------------------------------------
// GEMM: C[M,N](fp32) = A[M,K](bf16, row-major) * B[K,N](fp32, row-major).
// 256 threads = 8 waves; block tile 128x128, K-step 64, LDS double-buffered.
// Wave (wm 0..3, wn 0..1) owns a 32x64 tile = 2x4 WMMA accumulators.
// As[m][k] row-major; Bs[n][k] transposed, so both fragment types load with
// two ds_load_b128 per lane. Pitch 72 halfs keeps 16B alignment (144%16==0).
// ---------------------------------------------------------------------------
__global__ __launch_bounds__(256)
void gemm_bf16_kernel(const __bf16* __restrict__ A, const float* __restrict__ Bm,
                      float* __restrict__ C, int M, int N, int K)
{
  __shared__ alignas(16) __bf16 As[2][128][72];
  __shared__ alignas(16) __bf16 Bs[2][128][72];

  const int tid  = threadIdx.x;
  const int wave = tid >> 5;
  const int lane = tid & 31;
  const int lg   = lane >> 4;   // half-group (0/1)
  const int ln   = lane & 15;
  const int wm   = wave & 3;    // 4 M slots of 32 rows
  const int wn   = wave >> 2;   // 2 N slots of 64 cols
  const int bm   = blockIdx.y * 128;
  const int bn   = blockIdx.x * 128;

  const v8f vzero = {0.f,0.f,0.f,0.f,0.f,0.f,0.f,0.f};
  v8f acc[2][4];
#pragma unroll
  for (int i = 0; i < 2; ++i)
#pragma unroll
    for (int j = 0; j < 4; ++j) acc[i][j] = vzero;

  // Stage tile starting at column kt into buffer `buf`.
  auto stage = [&](int kt, int buf) {
    // A (bf16): 128 rows x 64 cols = 1024 16-byte chunks, async to LDS.
#pragma unroll
    for (int i = tid; i < 1024; i += 256) {
      const int r  = i >> 3;          // 8 chunks per row
      const int c8 = (i & 7) * 8;     // element offset within row
      async_ld16(&As[buf][r][c8],
                 A + (size_t)(bm + r) * K + kt + c8);
    }
    // B (fp32): 64 rows x 128 cols; convert + transposed store.
#pragma unroll
    for (int i = tid; i < 2048; i += 256) {
      const int r = i >> 5;
      const int c = (i & 31) << 2;
      const float* bp = Bm + (size_t)(kt + r) * N + bn + c;
      __builtin_prefetch(bp + (size_t)64 * N, 0, 0);
      const float4 f = *(const float4*)bp;
      Bs[buf][c+0][r] = (__bf16)f.x; Bs[buf][c+1][r] = (__bf16)f.y;
      Bs[buf][c+2][r] = (__bf16)f.z; Bs[buf][c+3][r] = (__bf16)f.w;
    }
  };

  stage(0, 0);
  const int niter = K >> 6;
  for (int it = 0; it < niter; ++it) {
    // Publish buffer it&1 (own async issues done + all waves' stores visible);
    // this barrier also retires iteration it-1's reads of buffer (it+1)&1.
    wait_async0();
    __syncthreads();
    if (it + 1 < niter) stage((it + 1) << 6, (it + 1) & 1);

    const int buf = it & 1;
#pragma unroll
    for (int s = 0; s < 2; ++s) {           // two K=32 WMMA steps
      v16bf af[2];
#pragma unroll
      for (int i = 0; i < 2; ++i) {
        const int m  = wm * 32 + i * 16 + ln;
        const int k0 = s * 32 + lg * 8;     // A: e0-7 at k0, e8-15 at k0+16
        af[i] = ld_frag(&As[buf][m][k0], &As[buf][m][k0 + 16]);
      }
      v16bf bfrag[4];
#pragma unroll
      for (int j = 0; j < 4; ++j) {
        const int n  = wn * 64 + j * 16 + ln;
        const int k0 = s * 32 + lg * 8;     // B: e0-7 at k0, e8-15 at k0+16
        bfrag[j] = ld_frag(&Bs[buf][n][k0], &Bs[buf][n][k0 + 16]);
      }
#pragma unroll
      for (int i = 0; i < 2; ++i)
#pragma unroll
        for (int j = 0; j < 4; ++j)
          acc[i][j] = wmma_bf16(af[i], bfrag[j], acc[i][j]);
    }
  }

  // C/D layout: row = vj + 8*lg (+tile base), col = ln (+tile base).
#pragma unroll
  for (int i = 0; i < 2; ++i)
#pragma unroll
    for (int j = 0; j < 4; ++j) {
      const int row = bm + wm * 32 + i * 16 + 8 * lg;
      const int col = bn + wn * 64 + j * 16 + ln;
#pragma unroll
      for (int vj = 0; vj < 8; ++vj)
        C[(size_t)(row + vj) * N + col] = acc[i][j][vj];
    }
}

// ---------------------------------------------------------------------------
// Flash attention with GQA (rep=2) and causal mask.
// grid = (S/64, NH, B), 128 threads = 4 waves; wave w owns q rows w*16..+15.
// Per KV tile: S = Q K^T via WMMA, online softmax (C-fragment rows live on
// 16-lane half-groups -> shfl_xor reductions), P -> LDS bf16, O += P V via
// WMMA with V staged transposed. Output written bf16 for the Wo GEMM.
// ---------------------------------------------------------------------------
__global__ __launch_bounds__(128)
void attn_kernel(const float* __restrict__ q, const float* __restrict__ k,
                 const float* __restrict__ v, __bf16* __restrict__ o)
{
  const int qb  = blockIdx.x;   // 64-row q block
  const int h   = blockIdx.y;
  const int b   = blockIdx.z;
  const int kvh = h >> 1;       // repeat_interleave(rep=2)

  __shared__ alignas(16) __bf16 Qs[64][72];
  __shared__ alignas(16) __bf16 Ks[64][72];      // Ks[kv][hd]
  __shared__ alignas(16) __bf16 Vt[64][72];      // Vt[hd][kv]
  __shared__ alignas(16) __bf16 Pw[4][16][72];   // per-wave P tile

  const int tid  = threadIdx.x;
  const int wave = tid >> 5;
  const int lane = tid & 31;
  const int lg   = lane >> 4;
  const int ln   = lane & 15;

  for (int i = tid; i < 1024; i += 128) {
    const int r = i >> 4;
    const int c = (i & 15) << 2;
    const float4 f = *(const float4*)
        (q + (((size_t)b * Sc + qb * 64 + r) * NHc + h) * HDc + c);
    Qs[r][c+0] = (__bf16)f.x; Qs[r][c+1] = (__bf16)f.y;
    Qs[r][c+2] = (__bf16)f.z; Qs[r][c+3] = (__bf16)f.w;
  }
  __syncthreads();

  v16bf qf[2];
#pragma unroll
  for (int s = 0; s < 2; ++s) {
    const int m  = wave * 16 + ln;
    const int k0 = s * 32 + lg * 8;
    qf[s] = ld_frag(&Qs[m][k0], &Qs[m][k0 + 16]);
  }

  const v8f vzero = {0.f,0.f,0.f,0.f,0.f,0.f,0.f,0.f};
  v8f of[4];
  float m_run[8], l_run[8];
#pragma unroll
  for (int j = 0; j < 4; ++j) of[j] = vzero;
#pragma unroll
  for (int vj = 0; vj < 8; ++vj) { m_run[vj] = -1e30f; l_run[vj] = 0.f; }

  for (int jt = 0; jt <= qb; ++jt) {
    __syncthreads();
    for (int i = tid; i < 1024; i += 128) {
      const int r = i >> 4;
      const int c = (i & 15) << 2;
      const size_t base = (((size_t)b * Sc + jt * 64 + r) * NKVc + kvh) * HDc + c;
      const float4 fk = *(const float4*)(k + base);
      Ks[r][c+0] = (__bf16)fk.x; Ks[r][c+1] = (__bf16)fk.y;
      Ks[r][c+2] = (__bf16)fk.z; Ks[r][c+3] = (__bf16)fk.w;
      const float4 fv = *(const float4*)(v + base);
      Vt[c+0][r] = (__bf16)fv.x; Vt[c+1][r] = (__bf16)fv.y;
      Vt[c+2][r] = (__bf16)fv.z; Vt[c+3][r] = (__bf16)fv.w;
    }
    __syncthreads();

    v8f sf[4];
#pragma unroll
    for (int j = 0; j < 4; ++j) sf[j] = vzero;
#pragma unroll
    for (int s = 0; s < 2; ++s)
#pragma unroll
      for (int j = 0; j < 4; ++j) {
        const int n  = j * 16 + ln;
        const int k0 = s * 32 + lg * 8;
        v16bf kf = ld_frag(&Ks[n][k0], &Ks[n][k0 + 16]);
        sf[j] = wmma_bf16(qf[s], kf, sf[j]);
      }

    const int qbase  = qb * 64 + wave * 16 + 8 * lg;
    const int kvbase = jt * 64 + ln;
#pragma unroll
    for (int j = 0; j < 4; ++j)
#pragma unroll
      for (int vj = 0; vj < 8; ++vj) {
        const float sv = sf[j][vj] * 0.125f;           // 1/sqrt(64)
        sf[j][vj] = ((kvbase + j * 16) <= (qbase + vj)) ? sv : -1e30f;
      }

    float escale[8], rsum[8];
#pragma unroll
    for (int vj = 0; vj < 8; ++vj) {
      float mx = fmaxf(fmaxf(sf[0][vj], sf[1][vj]), fmaxf(sf[2][vj], sf[3][vj]));
#pragma unroll
      for (int sh = 1; sh < 16; sh <<= 1)
        mx = fmaxf(mx, __shfl_xor(mx, sh, 32));
      const float mnew = fmaxf(m_run[vj], mx);
      escale[vj] = __expf(m_run[vj] - mnew);
      m_run[vj]  = mnew;
      rsum[vj]   = 0.f;
    }
#pragma unroll
    for (int j = 0; j < 4; ++j)
#pragma unroll
      for (int vj = 0; vj < 8; ++vj) {
        const float p = __expf(sf[j][vj] - m_run[vj]);
        sf[j][vj] = p;
        rsum[vj] += p;
      }
#pragma unroll
    for (int vj = 0; vj < 8; ++vj) {
#pragma unroll
      for (int sh = 1; sh < 16; sh <<= 1)
        rsum[vj] += __shfl_xor(rsum[vj], sh, 32);
      l_run[vj] = l_run[vj] * escale[vj] + rsum[vj];
    }
#pragma unroll
    for (int j = 0; j < 4; ++j)
#pragma unroll
      for (int vj = 0; vj < 8; ++vj)
        of[j][vj] *= escale[vj];

#pragma unroll
    for (int j = 0; j < 4; ++j)
#pragma unroll
      for (int vj = 0; vj < 8; ++vj)
        Pw[wave][vj + 8 * lg][j * 16 + ln] = (__bf16)sf[j][vj];
    __syncthreads();

#pragma unroll
    for (int s = 0; s < 2; ++s) {
      const int k0a = s * 32 + lg * 8;
      v16bf pf = ld_frag(&Pw[wave][ln][k0a], &Pw[wave][ln][k0a + 16]);
#pragma unroll
      for (int j = 0; j < 4; ++j) {
        const int n  = j * 16 + ln;            // hd column
        const int k0 = s * 32 + lg * 8;        // kv position
        v16bf vf = ld_frag(&Vt[n][k0], &Vt[n][k0 + 16]);
        of[j] = wmma_bf16(pf, vf, of[j]);
      }
    }
  }

  const int qrow = qb * 64 + wave * 16 + 8 * lg;
#pragma unroll
  for (int j = 0; j < 4; ++j)
#pragma unroll
    for (int vj = 0; vj < 8; ++vj)
      o[(((size_t)b * Sc + qrow + vj) * NHc + h) * HDc + j * 16 + ln] =
          (__bf16)(of[j][vj] / l_run[vj]);
}

// ---------------------------------------------------------------------------
// Elementwise helpers.
// ---------------------------------------------------------------------------
__global__ __launch_bounds__(256)
void embed_kernel(const int* __restrict__ ids, const float* __restrict__ emb,
                  float* __restrict__ h)
{
  const int row = blockIdx.x;
  const int tok = ids[row];
  const float4 f = *(const float4*)(emb + (size_t)tok * Hc + threadIdx.x * 4);
  *(float4*)(h + (size_t)row * Hc + threadIdx.x * 4) = f;
}

// RMSNorm: fp32 residual in, bf16 GEMM operand out.
__global__ __launch_bounds__(256)
void rmsnorm_kernel(const float* __restrict__ x, const float* __restrict__ w,
                    __bf16* __restrict__ y)
{
  __shared__ float red[256];
  const int row = blockIdx.x;
  const float4 f = *(const float4*)(x + (size_t)row * Hc + threadIdx.x * 4);
  red[threadIdx.x] = f.x*f.x + f.y*f.y + f.z*f.z + f.w*f.w;
  __syncthreads();
  for (int s = 128; s > 0; s >>= 1) {
    if (threadIdx.x < s) red[threadIdx.x] += red[threadIdx.x + s];
    __syncthreads();
  }
  const float inv = rsqrtf(red[0] * (1.0f / Hc) + 1e-6f);
  const float4 wv = *(const float4*)(w + threadIdx.x * 4);
  v4bf outv;
  outv[0] = (__bf16)(f.x * inv * wv.x); outv[1] = (__bf16)(f.y * inv * wv.y);
  outv[2] = (__bf16)(f.z * inv * wv.z); outv[3] = (__bf16)(f.w * inv * wv.w);
  *(v4bf*)(y + (size_t)row * Hc + threadIdx.x * 4) = outv;
}

__global__ __launch_bounds__(256)
void rope_kernel(float* __restrict__ x, int nheads)
{
  const int idx = blockIdx.x * 256 + threadIdx.x;
  const int d = idx & 31;           // rotation pair index
  const int t = idx >> 5;           // (b*S + s)*nheads + head
  const int pos = (t / nheads) % Sc;
  float* p = x + (size_t)t * HDc;
  const float inv = __powf(10000.0f, -((float)(2 * d)) * (1.0f / 64.0f));
  float sn, cs;
  __sincosf((float)pos * inv, &sn, &cs);
  const float x1 = p[d], x2 = p[d + 32];
  p[d]      = x1 * cs - x2 * sn;
  p[d + 32] = x2 * cs + x1 * sn;
}

// mid = silu(g) * u, emitted in bf16 for the Wd GEMM.
__global__ __launch_bounds__(256)
void silu_mul_kernel(const float* __restrict__ g, const float* __restrict__ u,
                     __bf16* __restrict__ out, int n)
{
  const int i = blockIdx.x * 256 + threadIdx.x;
  if (i < n) {
    const float gv = g[i];
    out[i] = (__bf16)(gv * (1.0f / (1.0f + __expf(-gv))) * u[i]);
  }
}

__global__ __launch_bounds__(256)
void add_kernel(float* __restrict__ a, const float* __restrict__ b, int n)
{
  const int i = blockIdx.x * 256 + threadIdx.x;
  if (i < n) a[i] += b[i];
}

// ---------------------------------------------------------------------------
// Host orchestration (all launches on `stream`; same stream => ordered).
// Workspace (112 MB): fp32 h(2M) q(2M) k(1M) v(1M) t0f(8M) t1f(8M);
// bf16 xnb(2M) aob(2M) t0b(8M).
// ---------------------------------------------------------------------------
extern "C" void kernel_launch(void* const* d_in, const int* in_sizes, int n_in,
                              void* d_out, int out_size, void* d_ws, size_t ws_size,
                              hipStream_t stream)
{
  (void)in_sizes; (void)n_in; (void)out_size; (void)ws_size;

  const int*   ids    = (const int*)  d_in[0];
  const float* embedw = (const float*)d_in[1];
  const float* Wq     = (const float*)d_in[2];
  const float* Wk     = (const float*)d_in[3];
  const float* Wv     = (const float*)d_in[4];
  const float* Wo     = (const float*)d_in[5];
  const float* Wg     = (const float*)d_in[6];
  const float* Wu     = (const float*)d_in[7];
  const float* Wd     = (const float*)d_in[8];
  const float* ln1    = (const float*)d_in[9];
  const float* ln2    = (const float*)d_in[10];
  const float* normw  = (const float*)d_in[11];
  const float* lmhead = (const float*)d_in[12];
  float* out = (float*)d_out;

  const int M = Bc * Sc;  // 2048 token rows

  float* ws  = (float*)d_ws;
  float* h   = ws;
  float* qb  = h   + (size_t)M * Hc;
  float* kb  = qb  + (size_t)M * Hc;
  float* vb  = kb  + (size_t)M * (Hc / 2);
  float* t0f = vb  + (size_t)M * (Hc / 2);
  float* t1f = t0f + (size_t)M * Ic;
  __bf16* xnb = (__bf16*)(t1f + (size_t)M * Ic);
  __bf16* aob = xnb + (size_t)M * Hc;
  __bf16* t0b = aob + (size_t)M * Hc;

  embed_kernel<<<M, 256, 0, stream>>>(ids, embedw, h);

  for (int l = 0; l < Lc; ++l) {
    const float* wq = Wq + (size_t)l * Hc * Hc;
    const float* wk = Wk + (size_t)l * Hc * (Hc / 2);
    const float* wv = Wv + (size_t)l * Hc * (Hc / 2);
    const float* wo = Wo + (size_t)l * Hc * Hc;
    const float* wg = Wg + (size_t)l * Hc * Ic;
    const float* wu = Wu + (size_t)l * Hc * Ic;
    const float* wd = Wd + (size_t)l * Ic * Hc;

    rmsnorm_kernel<<<M, 256, 0, stream>>>(h, ln1 + (size_t)l * Hc, xnb);
    gemm_bf16_kernel<<<dim3(Hc/128,     M/128), 256, 0, stream>>>(xnb, wq, qb, M, Hc,   Hc);
    gemm_bf16_kernel<<<dim3((Hc/2)/128, M/128), 256, 0, stream>>>(xnb, wk, kb, M, Hc/2, Hc);
    gemm_bf16_kernel<<<dim3((Hc/2)/128, M/128), 256, 0, stream>>>(xnb, wv, vb, M, Hc/2, Hc);
    rope_kernel<<<(M * NHc  * 32) / 256, 256, 0, stream>>>(qb, NHc);
    rope_kernel<<<(M * NKVc * 32) / 256, 256, 0, stream>>>(kb, NKVc);
    attn_kernel<<<dim3(Sc/64, NHc, Bc), 128, 0, stream>>>(qb, kb, vb, aob);
    gemm_bf16_kernel<<<dim3(Hc/128, M/128), 256, 0, stream>>>(aob, wo, t0f, M, Hc, Hc);
    add_kernel<<<(M * Hc) / 256, 256, 0, stream>>>(h, t0f, M * Hc);

    rmsnorm_kernel<<<M, 256, 0, stream>>>(h, ln2 + (size_t)l * Hc, xnb);
    gemm_bf16_kernel<<<dim3(Ic/128, M/128), 256, 0, stream>>>(xnb, wg, t0f, M, Ic, Hc);
    gemm_bf16_kernel<<<dim3(Ic/128, M/128), 256, 0, stream>>>(xnb, wu, t1f, M, Ic, Hc);
    silu_mul_kernel<<<(M * Ic) / 256, 256, 0, stream>>>(t0f, t1f, t0b, M * Ic);
    gemm_bf16_kernel<<<dim3(Hc/128, M/128), 256, 0, stream>>>(t0b, wd, t0f, M, Hc, Ic);
    add_kernel<<<(M * Hc) / 256, 256, 0, stream>>>(h, t0f, M * Hc);
  }

  rmsnorm_kernel<<<M, 256, 0, stream>>>(h, normw, xnb);
  gemm_bf16_kernel<<<dim3(Vc/128, M/128), 256, 0, stream>>>(xnb, lmhead, out, M, Vc, Hc);
}
